// NonLocalBlock_13108240188053
// MI455X (gfx1250) — matrix operand order repaired and checked by reading, not compiled, not run on hardware
//
#include <hip/hip_runtime.h>
#include <hip/hip_bf16.h>

typedef __bf16 bf16_t;
typedef __attribute__((ext_vector_type(16))) __bf16 v16bf;
typedef __attribute__((ext_vector_type(8)))  float   v8f;
typedef __attribute__((ext_vector_type(4)))  unsigned int v4u;
typedef __attribute__((ext_vector_type(8)))  int v8i;
typedef __attribute__((ext_vector_type(4)))  int v4i;

#define BATCH 4
#define CDIM  256
#define NDIM  4096
#define SCALE 0.0625f   // 1/sqrt(256)

// ---------------------------------------------------------------------------
// WMMA wrapper: D = A(16x32 bf16) * B(32x16 bf16) + C(16x16 f32)
// ---------------------------------------------------------------------------
__device__ __forceinline__ v8f wmma_bf16(v16bf a, v16bf b, v8f c) {
    return __builtin_amdgcn_wmma_f32_16x16x32_bf16(
        /*neg_a=*/false, a, /*neg_b=*/false, b,
        /*c_mod=*/(short)0, c, /*reuse_a=*/false, /*reuse_b=*/false);
}

// A fragment (16x32, M x K), source row-major [M][K], stride ld elements.
__device__ __forceinline__ v16bf load_a_frag(const bf16_t* p, int ld) {
    const int lane = threadIdx.x & 31;
    const int row  = lane & 15;
    const int half = lane >> 4;
    const bf16_t* r0 = p + row * ld + half * 8;
    const bf16_t* r1 = p + row * ld + 16 + half * 8;
    v16bf a;
#pragma unroll
    for (int e = 0; e < 8; ++e) a[e] = r0[e];
#pragma unroll
    for (int e = 0; e < 8; ++e) a[8 + e] = r1[e];
    return a;
}

// Same as above but converting from f32 source on the fly.
__device__ __forceinline__ v16bf load_a_frag_f32(const float* p, int ld) {
    const int lane = threadIdx.x & 31;
    const int row  = lane & 15;
    const int half = lane >> 4;
    const float* r0 = p + row * ld + half * 8;
    const float* r1 = p + row * ld + 16 + half * 8;
    v16bf a;
#pragma unroll
    for (int e = 0; e < 8; ++e) a[e] = (__bf16)r0[e];
#pragma unroll
    for (int e = 0; e < 8; ++e) a[8 + e] = (__bf16)r1[e];
    return a;
}

// B fragment (32x16, K x N), source stored N-major: [N][K], stride ldn.
__device__ __forceinline__ v16bf load_b_frag(const bf16_t* p, int ldn) {
    const int lane = threadIdx.x & 31;
    const bf16_t* r = p + (lane & 15) * ldn + (lane >> 4) * 8;
    v16bf b;
#pragma unroll
    for (int e = 0; e < 8; ++e) b[e] = r[e];
#pragma unroll
    for (int e = 0; e < 8; ++e) b[8 + e] = r[16 + e];
    return b;
}

// ---------------------------------------------------------------------------
// TDM: issue an async 2-D tile load global->LDS (D# per ISA ch.8).
// Element size 2B; LDS padding: pad_amount 4 DWORDs every 2^(iv+1) DWORDs.
// dims/strides in elements; gaddr is a byte address; wave-uniform arguments.
// Uses the 6-arg clang-23 builtin form (groups 2/3 + trailing group zeroed:
// 2-D tile, tile_dim2/3 = 0 => unused).
// ---------------------------------------------------------------------------
__device__ __forceinline__ void tdm_load_2d(unsigned lds_off, const void* gptr,
                                            unsigned tensor_d0, unsigned tensor_d1,
                                            unsigned tile_d0, unsigned tile_d1,
                                            unsigned stride0, unsigned iv_code) {
    const unsigned long long ga = (unsigned long long)(uintptr_t)gptr;
    v4u g0;
    g0[0] = 1u;                                    // count=1 (valid), user mode
    g0[1] = lds_off;                               // lds_addr
    g0[2] = (unsigned)ga;                          // global_addr[31:0]
    g0[3] = (unsigned)(ga >> 32) | 0x80000000u;    // addr[56:32] | type=2<<30
    v8i g1;
    g1[0] = (int)((1u << 16)                       // data_size = 2 bytes
                  | (1u << 20)                     // pad_enable
                  | (iv_code << 22)                // pad_interval code
                  | (3u << 25));                   // pad_amount = 4 DWORDs
    g1[1] = (int)(tensor_d0 << 16);                // tensor_dim0[15:0] @ bit48
    g1[2] = (int)((tensor_d0 >> 16) | (tensor_d1 << 16));
    g1[3] = (int)((tensor_d1 >> 16) | (tile_d0 << 16));
    g1[4] = (int)(tile_d1 & 0xFFFFu);              // tile_dim1 (tile_dim2=0)
    g1[5] = (int)stride0;                          // tensor_dim0_stride[31:0]
    g1[6] = 0;
    g1[7] = 0;
    const v4i z4 = {};
    const v8i z8 = {};
    __builtin_amdgcn_tensor_load_to_lds(g0, g1, z4, z4, z8, 0);
}

// ---------------------------------------------------------------------------
// Kernel 1: fused QKV projection.  out[o,n] = sum_c W[o,c] * x[c,n] + bias[o]
// grid: (NDIM/64, CDIM/64, BATCH*3); block 256 (8 waves, 4x2 wave tiles 16x32)
// Stores: s=0 -> Q token-major [n][c]; s=1 -> K token-major [n][c];
//         s=2 -> V channel-major [c][n]   (all bf16)
// ---------------------------------------------------------------------------
__global__ __launch_bounds__(256)
void qkv_kernel(const float* __restrict__ x,
                const float* __restrict__ wq, const float* __restrict__ bq,
                const float* __restrict__ wk, const float* __restrict__ bk,
                const float* __restrict__ wv, const float* __restrict__ bv,
                bf16_t* __restrict__ Qt, bf16_t* __restrict__ Kt,
                bf16_t* __restrict__ Vc) {
    __shared__ __align__(16) bf16_t Wt[64][40];   // [m][k] row-major
    __shared__ __align__(16) bf16_t Xt[64][40];   // [n][k] (transposed from x)

    const int b = blockIdx.z / 3;
    const int s = blockIdx.z % 3;
    const float* W    = (s == 0) ? wq : (s == 1) ? wk : wv;
    const float* bias = (s == 0) ? bq : (s == 1) ? bk : bv;

    const int n_blk = blockIdx.x * 64;
    const int m_blk = blockIdx.y * 64;
    const int tid   = threadIdx.x;
    const int wave  = tid >> 5;
    const int lane  = tid & 31;
    const int m_w   = (wave & 3) * 16;
    const int n_w   = (wave >> 2) * 32;

    const float* xb = x + (size_t)b * CDIM * NDIM;

    v8f d0 = {}, d1 = {};

    for (int kk = 0; kk < CDIM; kk += 32) {
        __syncthreads();
        {   // W tile 64(m) x 32(k)
            const int m  = tid >> 2;
            const int k0 = (tid & 3) * 8;
            const float* src = W + (size_t)(m_blk + m) * CDIM + kk + k0;
#pragma unroll
            for (int j = 0; j < 8; ++j) Wt[m][k0 + j] = (__bf16)src[j];
        }
        {   // x tile transposed -> Xt[n][k]
            const int k  = tid >> 3;
            const int n0 = (tid & 7) * 8;
            const float* src = xb + (size_t)(kk + k) * NDIM + n_blk + n0;
#pragma unroll
            for (int j = 0; j < 8; ++j) Xt[n0 + j][k] = (__bf16)src[j];
        }
        __syncthreads();

        v16bf a  = load_a_frag(&Wt[m_w][0], 40);
        v16bf b0 = load_b_frag(&Xt[n_w][0], 40);
        v16bf b1 = load_b_frag(&Xt[n_w + 16][0], 40);
        d0 = wmma_bf16(a, b0, d0);
        d1 = wmma_bf16(a, b1, d1);
    }

    const int col  = lane & 15;
    const int half = lane >> 4;
    bf16_t* Qb = Qt + (size_t)b * NDIM * CDIM;
    bf16_t* Kb = Kt + (size_t)b * NDIM * CDIM;
    bf16_t* Vb = Vc + (size_t)b * CDIM * NDIM;
#pragma unroll
    for (int r = 0; r < 8; ++r) {
        const int o  = m_blk + m_w + r + 8 * half;
        const float bo = bias[o];
        const int n0 = n_blk + n_w + col;
        const int n1 = n0 + 16;
        const float v0 = d0[r] + bo;
        const float v1 = d1[r] + bo;
        if (s == 0) {
            Qb[(size_t)n0 * CDIM + o] = (__bf16)v0;
            Qb[(size_t)n1 * CDIM + o] = (__bf16)v1;
        } else if (s == 1) {
            Kb[(size_t)n0 * CDIM + o] = (__bf16)v0;
            Kb[(size_t)n1 * CDIM + o] = (__bf16)v1;
        } else {
            Vb[(size_t)o * NDIM + n0] = (__bf16)v0;
            Vb[(size_t)o * NDIM + n1] = (__bf16)v1;
        }
    }
}

// ---------------------------------------------------------------------------
// Kernel 2: flash attention with TDM double buffering.  Each wave owns 16
// query rows; the block walks all 4096 keys in 32-token chunks.  K/V chunks
// are DMA'd into ping-pong LDS buffers by the Tensor Data Mover (issued one
// chunk ahead by wave 0, waited with s_wait_tensorcnt), so all 8 waves spend
// their issue slots on WMMAs.  Online softmax; logits never materialized.
// grid: (NDIM/128, BATCH); block 256 (8 waves)
// ---------------------------------------------------------------------------
__global__ __launch_bounds__(256)
void attn_kernel(const bf16_t* __restrict__ Qt, const bf16_t* __restrict__ Kt,
                 const bf16_t* __restrict__ Vc, bf16_t* __restrict__ Ot) {
    __shared__ __align__(16) bf16_t Klds[2][32][264];   // [j][c], 528B pitch
    __shared__ __align__(16) bf16_t Vlds[2][256][40];   // [c][j], 80B pitch
    __shared__ __align__(16) bf16_t Plds[8][16][32];    // per-wave P staging

    const int b    = blockIdx.y;
    const int tid  = threadIdx.x;
    const int wave = tid >> 5;
    const int lane = tid & 31;
    const int col  = lane & 15;
    const int half = lane >> 4;
    const int i0   = blockIdx.x * 128 + wave * 16;

    const bf16_t* Qb = Qt + (size_t)b * NDIM * CDIM;
    const bf16_t* Kb = Kt + (size_t)b * NDIM * CDIM;
    const bf16_t* Vb = Vc + (size_t)b * CDIM * NDIM;
    bf16_t*       Ob = Ot + (size_t)b * NDIM * CDIM;

    const unsigned koff0 = (unsigned)(uintptr_t)&Klds[0][0][0];
    const unsigned koff1 = (unsigned)(uintptr_t)&Klds[1][0][0];
    const unsigned voff0 = (unsigned)(uintptr_t)&Vlds[0][0][0];
    const unsigned voff1 = (unsigned)(uintptr_t)&Vlds[1][0][0];

    // Kick off the DMA for chunk 0 into buffer 0.
    if (wave == 0) {
        // K tile: 32 rows (j) x 256 (c); row = 128 DW -> pad_interval code 6
        tdm_load_2d(koff0, Kb, CDIM, NDIM, CDIM, 32, CDIM, 6);
        // V tile: 256 rows (c) x 32 (j); row = 16 DW -> pad_interval code 3
        tdm_load_2d(voff0, Vb, NDIM, CDIM, 32, CDIM, NDIM, 3);
    }

    // Cache the 16x256 Q slab as 8 A-fragments (stays in VGPRs all kernel).
    v16bf qf[8];
#pragma unroll
    for (int c8 = 0; c8 < 8; ++c8)
        qf[c8] = load_a_frag(Qb + (size_t)i0 * CDIM + c8 * 32, CDIM);

    v8f o[16];
#pragma unroll
    for (int t = 0; t < 16; ++t) o[t] = (v8f){};
    float m_r[8], l_r[8];
#pragma unroll
    for (int r = 0; r < 8; ++r) { m_r[r] = -1e30f; l_r[r] = 0.0f; }

#pragma unroll 1
    for (int j0 = 0; j0 < NDIM; j0 += 32) {
        const int cb = (j0 >> 5) & 1;
        if (wave == 0) {
            if (j0 + 32 < NDIM) {   // DMA next chunk into the other buffer
                const int jn = j0 + 32;
                tdm_load_2d(cb ? koff0 : koff1, Kb + (size_t)jn * CDIM,
                            CDIM, NDIM, CDIM, 32, CDIM, 6);
                tdm_load_2d(cb ? voff0 : voff1, Vb + jn,
                            NDIM, CDIM, 32, CDIM, NDIM, 3);
                __builtin_amdgcn_s_wait_tensorcnt(2);  // current pair landed
            } else {
                __builtin_amdgcn_s_wait_tensorcnt(0);
            }
        }
        __syncthreads();   // current buffer visible to all waves

        // S = Qt x K  (16 i x 32 j), contraction over C in 8 steps
        v8f s0 = {}, s1 = {};
#pragma unroll
        for (int c8 = 0; c8 < 8; ++c8) {
            v16bf k0 = load_b_frag(&Klds[cb][0][c8 * 32], 264);
            v16bf k1 = load_b_frag(&Klds[cb][16][c8 * 32], 264);
            s0 = wmma_bf16(qf[c8], k0, s0);
            s1 = wmma_bf16(qf[c8], k1, s1);
        }
#pragma unroll
        for (int r = 0; r < 8; ++r) { s0[r] *= SCALE; s1[r] *= SCALE; }

        // Row max over the chunk (reduce across 16-lane half-groups)
        float cm[8];
#pragma unroll
        for (int r = 0; r < 8; ++r) cm[r] = fmaxf(s0[r], s1[r]);
#pragma unroll
        for (int off = 1; off < 16; off <<= 1)
#pragma unroll
            for (int r = 0; r < 8; ++r)
                cm[r] = fmaxf(cm[r], __shfl_xor(cm[r], off, 32));

        float alpha[8];
#pragma unroll
        for (int r = 0; r < 8; ++r) {
            const float mn = fmaxf(m_r[r], cm[r]);
            alpha[r] = __expf(m_r[r] - mn);
            m_r[r]   = mn;
        }
        float rs[8];
#pragma unroll
        for (int r = 0; r < 8; ++r) {
            s0[r] = __expf(s0[r] - m_r[r]);
            s1[r] = __expf(s1[r] - m_r[r]);
            rs[r] = s0[r] + s1[r];
        }
#pragma unroll
        for (int off = 1; off < 16; off <<= 1)
#pragma unroll
            for (int r = 0; r < 8; ++r) rs[r] += __shfl_xor(rs[r], off, 32);
#pragma unroll
        for (int r = 0; r < 8; ++r) l_r[r] = l_r[r] * alpha[r] + rs[r];

        // Rescale running output accumulators
#pragma unroll
        for (int t = 0; t < 16; ++t)
#pragma unroll
            for (int r = 0; r < 8; ++r) o[t][r] *= alpha[r];

        // D-layout -> A-layout transpose of P through per-wave LDS
#pragma unroll
        for (int r = 0; r < 8; ++r) {
            Plds[wave][r + 8 * half][col]      = (__bf16)s0[r];
            Plds[wave][r + 8 * half][col + 16] = (__bf16)s1[r];
        }
        asm volatile("s_wait_dscnt 0" ::: "memory");   // cross-lane LDS dep
        v16bf pa = load_a_frag(&Plds[wave][0][0], 32);

        // O += P(16x32) x V^T(32x16 per c-tile), 16 c-tiles
#pragma unroll
        for (int t = 0; t < 16; ++t) {
            v16bf vb = load_b_frag(&Vlds[cb][t * 16][0], 40);
            o[t] = wmma_bf16(pa, vb, o[t]);
        }
        __syncthreads();   // reads done before this buffer is refilled
    }

    // Normalize and store token-major [i][c] (B-operand layout for kernel 3)
    float rinv[8];
#pragma unroll
    for (int r = 0; r < 8; ++r) rinv[r] = 1.0f / l_r[r];
#pragma unroll
    for (int t = 0; t < 16; ++t)
#pragma unroll
        for (int r = 0; r < 8; ++r) {
            const int i = i0 + r + 8 * half;
            Ob[(size_t)i * CDIM + t * 16 + col] = (__bf16)(o[t][r] * rinv[r]);
        }
}

// ---------------------------------------------------------------------------
// Kernel 3: projection + bias + residual.  y[o,n] = x[o,n] + bp[o] + Wp @ O
// grid: (NDIM/64, CDIM/64, BATCH); block 256 (8 waves, 16x32 wave tiles)
// ---------------------------------------------------------------------------
__global__ __launch_bounds__(256)
void proj_kernel(const float* __restrict__ x, const float* __restrict__ wp,
                 const float* __restrict__ bp, const bf16_t* __restrict__ Ot,
                 float* __restrict__ out) {
    const int b    = blockIdx.z;
    const int tid  = threadIdx.x;
    const int wave = tid >> 5;
    const int lane = tid & 31;
    const int col  = lane & 15;
    const int half = lane >> 4;
    const int m0   = blockIdx.y * 64 + (wave & 3) * 16;
    const int n0   = blockIdx.x * 64 + (wave >> 2) * 32;

    const bf16_t* Oblk = Ot + (size_t)b * NDIM * CDIM;
    v8f d0 = {}, d1 = {};
#pragma unroll
    for (int kk = 0; kk < CDIM; kk += 32) {
        v16bf a  = load_a_frag_f32(wp + (size_t)m0 * CDIM + kk, CDIM);
        v16bf b0 = load_b_frag(Oblk + (size_t)n0 * CDIM + kk, CDIM);
        v16bf b1 = load_b_frag(Oblk + (size_t)(n0 + 16) * CDIM + kk, CDIM);
        d0 = wmma_bf16(a, b0, d0);
        d1 = wmma_bf16(a, b1, d1);
    }

    const float* xb = x + (size_t)b * CDIM * NDIM;
    float* yb = out + (size_t)b * CDIM * NDIM;
#pragma unroll
    for (int r = 0; r < 8; ++r) {
        const int o  = m0 + r + 8 * half;
        const float bo = bp[o];
        const size_t base = (size_t)o * NDIM;
        const int na = n0 + col;
        const int nb = na + 16;
        yb[base + na] = xb[base + na] + d0[r] + bo;
        yb[base + nb] = xb[base + nb] + d1[r] + bo;
    }
}

// ---------------------------------------------------------------------------
extern "C" void kernel_launch(void* const* d_in, const int* in_sizes, int n_in,
                              void* d_out, int out_size, void* d_ws, size_t ws_size,
                              hipStream_t stream) {
    const float* x  = (const float*)d_in[0];
    const float* wq = (const float*)d_in[1];
    const float* bq = (const float*)d_in[2];
    const float* wk = (const float*)d_in[3];
    const float* bk = (const float*)d_in[4];
    const float* wv = (const float*)d_in[5];
    const float* bv = (const float*)d_in[6];
    const float* wp = (const float*)d_in[7];
    const float* bp = (const float*)d_in[8];
    float* out = (float*)d_out;

    char* ws = (char*)d_ws;
    const size_t slab = (size_t)BATCH * NDIM * CDIM * sizeof(bf16_t);  // 8 MB
    bf16_t* Qt = (bf16_t*)(ws);
    bf16_t* Kt = (bf16_t*)(ws + slab);
    bf16_t* Vc = (bf16_t*)(ws + 2 * slab);
    bf16_t* Ot = (bf16_t*)(ws + 3 * slab);

    qkv_kernel<<<dim3(NDIM / 64, CDIM / 64, BATCH * 3), dim3(256), 0, stream>>>(
        x, wq, bq, wk, bk, wv, bv, Qt, Kt, Vc);
    attn_kernel<<<dim3(NDIM / 128, BATCH), dim3(256), 0, stream>>>(Qt, Kt, Vc, Ot);
    proj_kernel<<<dim3(NDIM / 64, CDIM / 64, BATCH), dim3(256), 0, stream>>>(
        x, wp, bp, Ot, out);
}
